// CausalSelfAttention_30975304139020
// MI455X (gfx1250) — matrix operand order repaired and checked by reading, compiled, not run on hardware
//
#include <hip/hip_runtime.h>
#include <hip/hip_bf16.h>

typedef __attribute__((ext_vector_type(16))) _Float16 v16h;
typedef __attribute__((ext_vector_type(8)))  _Float16 v8h;
typedef __attribute__((ext_vector_type(8)))  float    v8f;
typedef __attribute__((ext_vector_type(4)))  unsigned int u32x4;
typedef __attribute__((ext_vector_type(8)))  unsigned int u32x8;

// ---------------- WMMA helpers ----------------

__device__ __forceinline__ v8f wmma16(v16h a, v16h b, v8f c) {
    // emits v_wmma_f32_16x16x32_f16
    return __builtin_amdgcn_wmma_f32_16x16x32_f16(
        /*neg_a=*/false, a, /*neg_b=*/false, b,
        /*c_mod=*/(short)0, c, /*reuse_a=*/false, /*reuse_b=*/false);
}

// A fragment 16x32 f16. lane&15 = row, (lane>>4)*16 = k base.
// halves 0..15 -> K = kb..kb+15.
__device__ __forceinline__ v16h load_a_frag(const _Float16* base, int ld,
                                            int row0, int k0, int lane) {
    int r  = row0 + (lane & 15);
    int kb = k0 + ((lane >> 4) << 4);
    const _Float16* p = base + (size_t)r * ld + kb;
    v8h lo = *(const v8h*)(p);
    v8h hi = *(const v8h*)(p + 8);
    v16h out;
#pragma unroll
    for (int j = 0; j < 8; ++j) { out[j] = lo[j]; out[j + 8] = hi[j]; }
    return out;
}

// B fragment 32x16 f16 from COLUMN-MAJOR global storage (col n's K contiguous).
__device__ __forceinline__ v16h load_b_frag(const _Float16* base, int ld,
                                            int col0, int k0, int lane) {
    int n  = col0 + (lane & 15);
    int kb = k0 + ((lane >> 4) << 4);
    const _Float16* p = base + (size_t)n * ld + kb;
    v8h lo = *(const v8h*)(p);
    v8h hi = *(const v8h*)(p + 8);
    v16h out;
#pragma unroll
    for (int j = 0; j < 8; ++j) { out[j] = lo[j]; out[j + 8] = hi[j]; }
    return out;
}

// B fragment from an LDS panel staged by TDM: 64 rows (cols of output),
// row stride 40 halves = 80B (16 DW data + 4 DW TDM pad -> bank-conflict-free).
#define PANEL_STRIDE_H 40
#define PANEL_BYTES    (64 * PANEL_STRIDE_H * 2)   // 5120 B per buffer

__device__ __forceinline__ v16h lds_b_frag(const _Float16* buf, int col0, int lane) {
    int n  = col0 + (lane & 15);
    int kb = (lane >> 4) << 4;
    const _Float16* p = buf + n * PANEL_STRIDE_H + kb;
    v8h lo = *(const v8h*)(p);       // ds_load_b128
    v8h hi = *(const v8h*)(p + 8);   // ds_load_b128
    v16h out;
#pragma unroll
    for (int j = 0; j < 8; ++j) { out[j] = lo[j]; out[j + 8] = hi[j]; }
    return out;
}

// ---------------- TDM: async 2D tile load Global->LDS ----------------
// D# per CDNA5 ISA: group0 (4 SGPR) + group1 (8 SGPR),
// VADDR2/3 = NULL -> 2D tensor. Tile = tile_d0 (contig k) x tile_d1 (rows).
// Padding: 4 DWORDs after every 16 DWORDs (one 64B row) -> 80B LDS row stride.

__device__ __forceinline__ unsigned lds_offset(const void* p) {
    return (unsigned)(unsigned long long)p;   // low 32 bits of generic ptr = LDS byte offset
}

__device__ __forceinline__ void tdm_load_2d_f16(
    unsigned lds_addr, const _Float16* gptr,
    unsigned tensor_d0, unsigned tensor_d1, unsigned stride0,
    unsigned tile_d0, unsigned tile_d1) {
    unsigned long long ga = (unsigned long long)gptr;
    u32x4 g0;
    g0[0] = 1u;                                          // count=1 (valid), user mode
    g0[1] = lds_addr;                                    // LDS byte address
    g0[2] = (unsigned)(ga & 0xFFFFFFFFu);                // global_addr[31:0]
    g0[3] = (unsigned)((ga >> 32) & 0x01FFFFFFu)         // global_addr[56:32]
          | (2u << 30);                                  // type=2 (image)
    u32x8 g1;
    g1[0] = (1u << 16)                                   // data_size = 2 bytes
          | (1u << 20)                                   // pad_enable
          | (3u << 22)                                   // pad_interval: 16 DWORDs
          | (3u << 25);                                  // pad_amount: 4 DWORDs
    g1[1] = (tensor_d0 & 0xFFFFu) << 16;                 // tensor_dim0[15:0]
    g1[2] = (tensor_d0 >> 16) | ((tensor_d1 & 0xFFFFu) << 16);
    g1[3] = (tensor_d1 >> 16) | ((tile_d0 & 0xFFFFu) << 16);
    g1[4] = (tile_d1 & 0xFFFFu);                         // tile_dim1 (tile_dim2=0)
    g1[5] = stride0;                                     // tensor_dim0_stride[31:0]
    g1[6] = 0u;
    g1[7] = 0u;
    asm volatile("tensor_load_to_lds %0, %1" :: "s"(g0), "s"(g1) : "memory");
}

// ---------------- Problem constants ----------------
#define TB   4
#define TT   2048
#define TC   1024
#define TH   16
#define TD   64
#define MROWS (TB * TT)          // 8192
#define C3    (3 * TC)           // 3072

// ---------------- Kernel 1: convert / transpose weights ----------------

__global__ __launch_bounds__(256) void cvt_kernel(
    const float* __restrict__ x, const float* __restrict__ wa,
    const float* __restrict__ wp,
    _Float16* __restrict__ xh, _Float16* __restrict__ waT,
    _Float16* __restrict__ wpT) {
    size_t i = (size_t)blockIdx.x * 256 + threadIdx.x;
    const size_t NX = (size_t)MROWS * TC;
    const size_t NA = (size_t)TC * C3;
    const size_t NP = (size_t)TC * TC;
    if (i < NX) xh[i] = (_Float16)x[i];
    if (i < NA) {
        size_t k = i / C3, n = i % C3;
        waT[n * TC + k] = (_Float16)wa[i];     // [3072][1024]
    }
    if (i < NP) {
        size_t k = i / TC, n = i % TC;
        wpT[n * TC + k] = (_Float16)wp[i];     // [1024][1024]
    }
}

// ---------------- Kernel 2: QKV GEMM with TDM-staged B panel ----------------
// Block = 8 waves = 128 output rows; panel = 64 output cols shared via LDS.
// Grid: (8192/128) * (3072/64) = 64*48 = 3072 blocks.

__global__ __launch_bounds__(256) void qkv_gemm_kernel(
    const _Float16* __restrict__ xh, const _Float16* __restrict__ waT,
    _Float16* __restrict__ Qb, _Float16* __restrict__ Kb,
    _Float16* __restrict__ Vt) {
    __shared__ __align__(16) _Float16 bpanel[2][64 * PANEL_STRIDE_H];
    int wv = threadIdx.x >> 5;
    int lane = threadIdx.x & 31;
    int m0 = (blockIdx.x / 48) * 128 + wv * 16;
    int n0 = (blockIdx.x % 48) * 64;
    const _Float16* wa_tile = waT + (size_t)n0 * TC;     // panel row 0, k 0

    if (wv == 0)
        tdm_load_2d_f16(lds_offset(&bpanel[0][0]), wa_tile, TC, C3, TC, 32, 64);

    v8f acc[4] = {};
    for (int k0 = 0; k0 < TC; k0 += 32) {
        int cur = (k0 >> 5) & 1;
        if (wv == 0) __builtin_amdgcn_s_wait_tensorcnt(0);   // buf[cur] landed
        __syncthreads();                                     // publish to all waves
        if (wv == 0 && k0 + 32 < TC)                         // overlap next DMA
            tdm_load_2d_f16(lds_offset(&bpanel[cur ^ 1][0]),
                            wa_tile + (k0 + 32), TC, C3, TC, 32, 64);
        v16h a = load_a_frag(xh, TC, m0, k0, lane);
#pragma unroll
        for (int j = 0; j < 4; ++j) {
            v16h b = lds_b_frag(&bpanel[cur][0], j * 16, lane);
            acc[j] = wmma16(a, b, acc[j]);
        }
        __syncthreads();                                     // reads done before reuse
    }

    int rhalf = (lane >> 4) << 3;
#pragma unroll
    for (int j = 0; j < 4; ++j) {
        int n   = n0 + j * 16 + (lane & 15);
        int sec = n / TC;                      // 0=q, 1=k, 2=v
        int cc  = n % TC;
        int h   = cc / TD, d = cc % TD;
#pragma unroll
        for (int v = 0; v < 8; ++v) {
            int m  = m0 + rhalf + v;
            int bb = m >> 11, t = m & (TT - 1);
            size_t bh = (size_t)(bb * TH + h);
            _Float16 val = (_Float16)acc[j][v];
            if (sec == 0)      Qb[(bh * TT + t) * TD + d] = val;
            else if (sec == 1) Kb[(bh * TT + t) * TD + d] = val;
            else               Vt[(bh * TD + d) * TT + t] = val;  // transposed
        }
    }
}

// ---------------- Kernel 3: flash attention, one wave per 16-query tile ----
// Grid: 64 heads * 128 q-tiles = 8192 blocks of 32 threads.

__global__ __launch_bounds__(32) void attn_kernel(
    const _Float16* __restrict__ Qb, const _Float16* __restrict__ Kb,
    const _Float16* __restrict__ Vt, _Float16* __restrict__ yh) {
    __shared__ __align__(16) _Float16 pbuf[16 * 32];
    int lane = threadIdx.x & 31;
    int bh = blockIdx.x >> 7;
    int qt = blockIdx.x & 127;
    int q0 = qt * 16;
    const _Float16* Qh = Qb + (size_t)bh * TT * TD;
    const _Float16* Kh = Kb + (size_t)bh * TT * TD;
    const _Float16* Vh = Vt + (size_t)bh * TD * TT;

    v16h qf0 = load_a_frag(Qh, TD, q0, 0, lane);
    v16h qf1 = load_a_frag(Qh, TD, q0, 32, lane);

    float m_i[8], l_i[8];
    v8f out[4] = {};
#pragma unroll
    for (int v = 0; v < 8; ++v) { m_i[v] = -__builtin_inff(); l_i[v] = 0.f; }

    const float scale = 0.125f;                // 1/sqrt(64)
    int col   = lane & 15;
    int rhalf = (lane >> 4) << 3;

    for (int kv = 0; kv < q0 + 16; kv += 32) {
        v8f s0 = {}, s1 = {};
        s0 = wmma16(qf0, load_b_frag(Kh, TD, kv,      0,  lane), s0);
        s0 = wmma16(qf1, load_b_frag(Kh, TD, kv,      32, lane), s0);
        s1 = wmma16(qf0, load_b_frag(Kh, TD, kv + 16, 0,  lane), s1);
        s1 = wmma16(qf1, load_b_frag(Kh, TD, kv + 16, 32, lane), s1);

#pragma unroll
        for (int v = 0; v < 8; ++v) {
            int q = q0 + rhalf + v;
            float a0 = s0[v] * scale;
            float a1 = s1[v] * scale;
            if (kv + col      > q) a0 = -__builtin_inff();
            if (kv + 16 + col > q) a1 = -__builtin_inff();

            float mx = fmaxf(a0, a1);
            mx = fmaxf(mx, __shfl_xor(mx, 1, 16));
            mx = fmaxf(mx, __shfl_xor(mx, 2, 16));
            mx = fmaxf(mx, __shfl_xor(mx, 4, 16));
            mx = fmaxf(mx, __shfl_xor(mx, 8, 16));

            float mnew  = fmaxf(m_i[v], mx);
            float alpha = __expf(m_i[v] - mnew);
            float p0 = __expf(a0 - mnew);
            float p1 = __expf(a1 - mnew);

            float rs = p0 + p1;
            rs += __shfl_xor(rs, 1, 16);
            rs += __shfl_xor(rs, 2, 16);
            rs += __shfl_xor(rs, 4, 16);
            rs += __shfl_xor(rs, 8, 16);

            l_i[v] = alpha * l_i[v] + rs;
            m_i[v] = mnew;
#pragma unroll
            for (int j = 0; j < 4; ++j) out[j][v] *= alpha;

            pbuf[(rhalf + v) * 32 + col]      = (_Float16)p0;
            pbuf[(rhalf + v) * 32 + 16 + col] = (_Float16)p1;
        }
        asm volatile("s_wait_dscnt 0" ::: "memory");

        v16h pf;
        {
            int r  = lane & 15;
            int kb = (lane >> 4) << 4;
            v8h lo = *(const v8h*)&pbuf[r * 32 + kb];
            v8h hi = *(const v8h*)&pbuf[r * 32 + kb + 8];
#pragma unroll
            for (int j = 0; j < 8; ++j) { pf[j] = lo[j]; pf[j + 8] = hi[j]; }
        }
        asm volatile("s_wait_dscnt 0" ::: "memory");

#pragma unroll
        for (int j = 0; j < 4; ++j) {
            v16h vb = load_b_frag(Vh, TT, j * 16, kv, lane);
            out[j] = wmma16(pf, vb, out[j]);
        }
    }

    int b = bh >> 4, h = bh & 15;
#pragma unroll
    for (int j = 0; j < 4; ++j) {
        int d = j * 16 + col;
#pragma unroll
        for (int v = 0; v < 8; ++v) {
            int q = q0 + rhalf + v;
            float val = out[j][v] / l_i[v];
            yh[((size_t)(b * TT + q)) * TC + h * TD + d] = (_Float16)val;
        }
    }
}

// ---------------- Kernel 4: projection GEMM with TDM-staged B ----------------
// Grid: (8192/128) * (1024/64) = 64*16 = 1024 blocks.

__global__ __launch_bounds__(256) void proj_gemm_kernel(
    const _Float16* __restrict__ yh, const _Float16* __restrict__ wpT,
    float* __restrict__ out) {
    __shared__ __align__(16) _Float16 bpanel[2][64 * PANEL_STRIDE_H];
    int wv = threadIdx.x >> 5;
    int lane = threadIdx.x & 31;
    int m0 = (blockIdx.x >> 4) * 128 + wv * 16;
    int n0 = (blockIdx.x & 15) * 64;
    const _Float16* wp_tile = wpT + (size_t)n0 * TC;

    if (wv == 0)
        tdm_load_2d_f16(lds_offset(&bpanel[0][0]), wp_tile, TC, TC, TC, 32, 64);

    v8f acc[4] = {};
    for (int k0 = 0; k0 < TC; k0 += 32) {
        int cur = (k0 >> 5) & 1;
        if (wv == 0) __builtin_amdgcn_s_wait_tensorcnt(0);
        __syncthreads();
        if (wv == 0 && k0 + 32 < TC)
            tdm_load_2d_f16(lds_offset(&bpanel[cur ^ 1][0]),
                            wp_tile + (k0 + 32), TC, TC, TC, 32, 64);
        v16h a = load_a_frag(yh, TC, m0, k0, lane);
#pragma unroll
        for (int j = 0; j < 4; ++j) {
            v16h b = lds_b_frag(&bpanel[cur][0], j * 16, lane);
            acc[j] = wmma16(a, b, acc[j]);
        }
        __syncthreads();
    }

    int rhalf = (lane >> 4) << 3;
#pragma unroll
    for (int j = 0; j < 4; ++j) {
        int n = n0 + j * 16 + (lane & 15);
#pragma unroll
        for (int v = 0; v < 8; ++v) {
            int m = m0 + rhalf + v;
            out[(size_t)m * TC + n] = acc[j][v];
        }
    }
}

// ---------------- Host launcher ----------------

extern "C" void kernel_launch(void* const* d_in, const int* in_sizes, int n_in,
                              void* d_out, int out_size, void* d_ws, size_t ws_size,
                              hipStream_t stream) {
    const float* x  = (const float*)d_in[0];
    const float* wa = (const float*)d_in[1];
    const float* wp = (const float*)d_in[2];
    float* out = (float*)d_out;

    char* ws = (char*)d_ws;
    size_t off = 0;
    _Float16* xh  = (_Float16*)(ws + off); off += (size_t)MROWS * TC * 2;
    _Float16* waT = (_Float16*)(ws + off); off += (size_t)C3 * TC * 2;
    _Float16* wpT = (_Float16*)(ws + off); off += (size_t)TC * TC * 2;
    _Float16* Qb  = (_Float16*)(ws + off); off += (size_t)MROWS * TC * 2;
    _Float16* Kb  = (_Float16*)(ws + off); off += (size_t)MROWS * TC * 2;
    _Float16* Vt  = (_Float16*)(ws + off); off += (size_t)MROWS * TC * 2;
    _Float16* yh  = (_Float16*)(ws + off); off += (size_t)MROWS * TC * 2;
    (void)ws_size; (void)n_in; (void)in_sizes; (void)out_size;

    cvt_kernel<<<(MROWS * TC + 255) / 256, 256, 0, stream>>>(x, wa, wp, xh, waT, wpT);
    qkv_gemm_kernel<<<64 * 48, 256, 0, stream>>>(xh, waT, Qb, Kb, Vt);
    attn_kernel<<<64 * 128, 32, 0, stream>>>(Qb, Kb, Vt, yh);
    proj_gemm_kernel<<<64 * 16, 256, 0, stream>>>(yh, wpT, out);
}
